// RRNetModel_1795296329750
// MI455X (gfx1250) — matrix operand order repaired and checked by reading, not compiled, not run on hardware
//
#include <hip/hip_runtime.h>
#include <hip/hip_bf16.h>
#include <cmath>

// ---------------- types ----------------
typedef __bf16 v16bf __attribute__((ext_vector_type(16)));
typedef __bf16 v8bf  __attribute__((ext_vector_type(8)));
typedef float  v8f   __attribute__((ext_vector_type(8)));

__device__ __forceinline__ unsigned short f2bf(float f) {
    unsigned int u = __float_as_uint(f);
    unsigned int r = u + 0x7FFFu + ((u >> 16) & 1u);   // RNE
    return (unsigned short)(r >> 16);
}
__device__ __forceinline__ float sigm(float x) { return 1.0f / (1.0f + __expf(-x)); }

// ---------------- generic bf16 WMMA GEMM ----------------
// C[M x N] = A[M x K] * W[N x K]^T (+ optional second A1/W1 pair) + bias
// A row-major stride K, W row-major stride K (so B-fragment columns are
// contiguous weight rows -> contiguous 16B loads, no transpose needed).
// EPI: 0 = fp32 out (+bias);  1 = BatchNorm(eval)+tanh -> bf16 out.
// NTW: N-tiles (16 cols each) per wave; A fragment reused NTW times.
template <int EPI, int NTW>
__global__ __launch_bounds__(128) void gemm_bf16_wmma(
    const unsigned short* __restrict__ A0, const unsigned short* __restrict__ W0, int K0,
    const unsigned short* __restrict__ A1, const unsigned short* __restrict__ W1, int K1,
    const float* __restrict__ bias,
    float* __restrict__ Cf, unsigned short* __restrict__ Cb, int ldc,
    const float* __restrict__ e_g, const float* __restrict__ e_b, float e_s)
{
    const int lane  = threadIdx.x & 31;
    const int wave  = threadIdx.x >> 5;
    const int half  = lane >> 4;    // which 16-lane group
    const int l16   = lane & 15;
    const int mrow  = blockIdx.y * 16;
    const int ncol0 = (blockIdx.x * 4 + wave) * (16 * NTW);

    v8f acc[NTW];
#pragma unroll
    for (int tn = 0; tn < NTW; ++tn) acc[tn] = (v8f){0.f,0.f,0.f,0.f,0.f,0.f,0.f,0.f};

    for (int pass = 0; pass < 2; ++pass) {
        const unsigned short* A = pass ? A1 : A0;
        const unsigned short* W = pass ? W1 : W0;
        const int K = pass ? K1 : K0;
        if (A == nullptr) continue;
        const unsigned short* arow = A + (size_t)(mrow + l16) * K;
        const unsigned short* wrow[NTW];
#pragma unroll
        for (int tn = 0; tn < NTW; ++tn)
            wrow[tn] = W + (size_t)(ncol0 + tn * 16 + l16) * K;

        for (int k0 = 0; k0 < K; k0 += 32) {
            // A 16x32 fragment: lanes 0-15 row M=l16 K{0..7,16..23}; lanes 16-31 K{8..15,24..31}
            v8bf a_lo = *(const v8bf*)(arow + k0 + half * 8);
            v8bf a_hi = *(const v8bf*)(arow + k0 + 16 + half * 8);
            v16bf af  = __builtin_shufflevector(a_lo, a_hi,
                                                0,1,2,3,4,5,6,7,8,9,10,11,12,13,14,15);
            if (NTW > 1 && k0 + 256 < K) {
                // streaming-weight path (pred/dec): hint ~8 k-steps ahead (global_prefetch)
#pragma unroll
                for (int tn = 0; tn < NTW; ++tn)
                    __builtin_prefetch(wrow[tn] + k0 + 256, 0, 1);
            }
#pragma unroll
            for (int tn = 0; tn < NTW; ++tn) {
                // B 32x16 fragment: lanes 0-15 N=l16 K{0..7,16..23}; lanes 16-31 K{8..15,24..31}
                v8bf b_lo = *(const v8bf*)(wrow[tn] + k0 + half * 8);
                v8bf b_hi = *(const v8bf*)(wrow[tn] + k0 + 16 + half * 8);
                v16bf bf  = __builtin_shufflevector(b_lo, b_hi,
                                                    0,1,2,3,4,5,6,7,8,9,10,11,12,13,14,15);
                acc[tn] = __builtin_amdgcn_wmma_f32_16x16x32_bf16(false, af, false, bf,
                                                                  (short)0, acc[tn], false, false);
            }
        }
    }

#pragma unroll
    for (int tn = 0; tn < NTW; ++tn) {
        const int n = ncol0 + tn * 16 + l16;
        const float bv = bias ? bias[n] : 0.f;
#pragma unroll
        for (int i = 0; i < 8; ++i) {
            const int m = mrow + half * 8 + i;   // C/D layout: VGPR i -> M = i (+8 hi half)
            float v = acc[tn][i] + bv;
            if (EPI == 0) {
                Cf[(size_t)m * ldc + n] = v;
            } else {                              // BN(eval: rm=0, rv=1) + tanh -> bf16
                v = tanhf(v * e_s * e_g[n] + e_b[n]);
                Cb[(size_t)m * ldc + n] = f2bf(v);
            }
        }
    }
}

// ---------------- small helper kernels ----------------
__global__ void cvt_bf16_kernel(const float* __restrict__ in, unsigned short* __restrict__ out, int n) {
    int i = blockIdx.x * blockDim.x + threadIdx.x;
    if (i < n) out[i] = f2bf(in[i]);
}
__global__ void add2_kernel(const float* a, const float* b, float* out, int n) {
    int i = blockIdx.x * blockDim.x + threadIdx.x;
    if (i < n) out[i] = a[i] + b[i];
}
__global__ void copyf_kernel(const float* a, float* out, int n) {
    int i = blockIdx.x * blockDim.x + threadIdx.x;
    if (i < n) out[i] = a[i];
}
__global__ void embed_kernel(const float* __restrict__ embW, const int* __restrict__ tokens,
                             unsigned short* __restrict__ out, int rows, int NI) {
    int i = blockIdx.x * blockDim.x + threadIdx.x;
    if (i >= rows * NI) return;
    int r = i / NI, c = i - r * NI;
    out[i] = f2bf(embW[(size_t)tokens[r] * NI + c]);
}
__global__ void init_state_kernel(const float* hx, const float* cx,
                                  float* h0, float* c0, float* h1, float* c1,
                                  unsigned short* h0b, unsigned short* h1b,
                                  float* stk, int* slen) {
    int i = blockIdx.x * blockDim.x + threadIdx.x;   // 32*1024
    if (i < 32 * 1024) {
        float hv0 = hx[i], hv1 = hx[32 * 1024 + i];
        h0[i] = hv0; h1[i] = hv1; c0[i] = cx[i]; c1[i] = cx[32 * 1024 + i];
        h0b[i] = f2bf(hv0); h1b[i] = f2bf(hv1);
        stk[i] = 0.f; stk[32 * 1024 + i] = 0.f;
        if (i < 32) slen[i] = 0;
    }
}
// gates in g: [i | f | g | o], bias already added by GEMM epilogue
__global__ void lstm_cell_kernel(const float* __restrict__ g, const float* __restrict__ c_prev,
                                 float* __restrict__ h, float* __restrict__ c,
                                 unsigned short* __restrict__ hb) {
    const int H = 1024;
    int i = blockIdx.x * blockDim.x + threadIdx.x;   // 32*1024
    if (i >= 32 * H) return;
    int b = i / H, j = i - b * H;
    const float* gr = g + (size_t)b * 4 * H;
    float gi = gr[j], gf = gr[H + j], gg = gr[2 * H + j], go = gr[3 * H + j];
    float c2 = sigm(gf) * c_prev[i] + sigm(gi) * tanhf(gg);
    float hv = sigm(go) * tanhf(c2);
    c[i] = c2; h[i] = hv;
    if (hb) hb[i] = f2bf(hv);
}
// state = [htm1 | si | hi] as bf16; also keep si in fp32 for the merge path
__global__ void build_state_kernel(const float* __restrict__ h1p, const float* __restrict__ ht0,
                                   const float* __restrict__ stk, const int* __restrict__ slen,
                                   unsigned short* __restrict__ state, float* __restrict__ si_f) {
    const int H = 1024;
    int i = blockIdx.x * blockDim.x + threadIdx.x;   // 32*3072
    if (i >= 32 * 3 * H) return;
    int b = i / (3 * H), col = i - b * (3 * H);
    float v;
    if (col < H) {
        v = h1p[(size_t)b * H + col];
    } else if (col < 2 * H) {
        int sl = slen[b];
        int top = sl - 1; if (top < 0) top = 0; if (top > 1) top = 1;
        v = (sl > 0) ? stk[((size_t)b * 2 + top) * H + (col - H)] : 0.f;
        si_f[(size_t)b * H + (col - H)] = v;
    } else {
        v = ht0[(size_t)b * H + (col - 2 * H)];
    }
    state[i] = f2bf(v);
}

__device__ __forceinline__ float block_reduce_sum(float v, float* red, int tid) {
    red[tid] = v; __syncthreads();
    for (int o = 128; o > 0; o >>= 1) { if (tid < o) red[tid] += red[tid + o]; __syncthreads(); }
    float r = red[0]; __syncthreads();
    return r;
}

// per batch-row: LayerNorm+tanh heads, masked argmax policy, value, merge/push, stack update
__global__ __launch_bounds__(256) void finalize_kernel(
    const float* __restrict__ y,           // [32 x 1024] : pi cols 0..511, vf cols 512..1023
    const float* __restrict__ hr, const float* __restrict__ cr, const float* __restrict__ si,
    const float* __restrict__ pi_lng, const float* __restrict__ pi_lnb,
    const float* __restrict__ pi_w2, const float* __restrict__ pi_b2,
    const float* __restrict__ vf_lng, const float* __restrict__ vf_lnb,
    const float* __restrict__ vf_w2, const float* __restrict__ vf_b2,
    float* __restrict__ stk, int* __restrict__ slen,
    float* __restrict__ h1n, float* __restrict__ c1n, unsigned short* __restrict__ h1nb,
    unsigned short* __restrict__ seq_bf, float* __restrict__ out_v, int t)
{
    const int H = 1024, NIv = 512;
    const int b = blockIdx.x, tid = threadIdx.x;
    __shared__ float red[256];
    __shared__ int sh_a;

    const float* yp = y + (size_t)b * 1024;
    const float* yv = yp + 512;

    // --- policy head: LN -> tanh -> 3 logits
    float s = 0.f;
    for (int j = tid; j < NIv; j += 256) s += yp[j];
    float mean = block_reduce_sum(s, red, tid) * (1.f / NIv);
    float vv = 0.f;
    for (int j = tid; j < NIv; j += 256) { float d = yp[j] - mean; vv += d * d; }
    float inv = rsqrtf(block_reduce_sum(vv, red, tid) * (1.f / NIv) + 1e-5f);
    float l0a = 0.f, l1a = 0.f, l2a = 0.f;
    for (int j = tid; j < NIv; j += 256) {
        float tj = tanhf((yp[j] - mean) * inv * pi_lng[j] + pi_lnb[j]);
        l0a += tj * pi_w2[j]; l1a += tj * pi_w2[NIv + j]; l2a += tj * pi_w2[2 * NIv + j];
    }
    l0a = block_reduce_sum(l0a, red, tid);
    l1a = block_reduce_sum(l1a, red, tid);
    l2a = block_reduce_sum(l2a, red, tid);

    // --- value head
    s = 0.f;
    for (int j = tid; j < NIv; j += 256) s += yv[j];
    float mean2 = block_reduce_sum(s, red, tid) * (1.f / NIv);
    vv = 0.f;
    for (int j = tid; j < NIv; j += 256) { float d = yv[j] - mean2; vv += d * d; }
    float inv2 = rsqrtf(block_reduce_sum(vv, red, tid) * (1.f / NIv) + 1e-5f);
    float va = 0.f;
    for (int j = tid; j < NIv; j += 256) {
        float tj = tanhf((yv[j] - mean2) * inv2 * vf_lng[j] + vf_lnb[j]);
        va += tj * vf_w2[j];
    }
    va = block_reduce_sum(va, red, tid);

    const int sl = slen[b];                 // all threads load BEFORE the barrier below
    if (tid == 0) {
        float lg0 = l0a + pi_b2[0], lg1 = l1a + pi_b2[1], lg2 = l2a + pi_b2[2];
        if (sl >= 2) lg0 = -1e30f;          // push masked
        if (sl == 0) lg2 = -1e30f;          // merge masked
        int a = 0;                          // jnp.argmax tie-break: lowest index
        if (lg1 > lg0) a = 1;
        if (lg2 > ((a == 1) ? lg1 : lg0)) a = 2;
        sh_a = a;
        out_v[(size_t)t * 32 + b] = va + vf_b2[0];
    }
    __syncthreads();
    const int a = sh_a;
    const bool push = (a == 0) && (sl < 2);
    for (int j = tid; j < H; j += 256) {
        float hrv = hr[(size_t)b * H + j];
        float ns = (a == 2) ? (0.5f * hrv + 0.5f * si[(size_t)b * H + j]) : hrv;
        h1n[(size_t)b * H + j] = ns;
        h1nb[(size_t)b * H + j] = f2bf(ns);
        c1n[(size_t)b * H + j] = cr[(size_t)b * H + j];
        seq_bf[((size_t)t * 32 + b) * H + j] = f2bf(ns);
        if (push) stk[((size_t)b * 2 + sl) * H + j] = hrv;
    }
    if (tid == 0) slen[b] = sl + (push ? 1 : 0) - (a == 2 ? 1 : 0);
}

__global__ void copy_hc_kernel(const float* h0, const float* h1, const float* c0, const float* c1,
                               float* out_h, float* out_c) {
    int i = blockIdx.x * blockDim.x + threadIdx.x;   // 32*1024
    if (i < 32 * 1024) {
        out_h[i] = h0[i]; out_h[32 * 1024 + i] = h1[i];
        out_c[i] = c0[i]; out_c[32 * 1024 + i] = c1[i];
    }
}

// ---------------- host ----------------
static inline int cdiv(int a, int b) { return (a + b - 1) / b; }

extern "C" void kernel_launch(void* const* d_in, const int* in_sizes, int n_in,
                              void* d_out, int out_size, void* d_ws, size_t ws_size,
                              hipStream_t stream) {
    (void)in_sizes; (void)n_in; (void)out_size; (void)ws_size;
    const int T = 128, B = 32, H = 1024, NI = 512, NT = 32000;

    const float* emb_W  = (const float*)d_in[0];
    const float* l0_wih = (const float*)d_in[1];
    const float* l0_whh = (const float*)d_in[2];
    const float* l0_bih = (const float*)d_in[3];
    const float* l0_bhh = (const float*)d_in[4];
    const float* rc_wih = (const float*)d_in[5];
    const float* rc_whh = (const float*)d_in[6];
    const float* rc_bih = (const float*)d_in[7];
    const float* rc_bhh = (const float*)d_in[8];
    const float* pi_w1  = (const float*)d_in[9];
    const float* pi_b1  = (const float*)d_in[10];
    const float* pi_lng = (const float*)d_in[11];
    const float* pi_lnb = (const float*)d_in[12];
    const float* pi_w2  = (const float*)d_in[13];
    const float* pi_b2  = (const float*)d_in[14];
    const float* vf_w1  = (const float*)d_in[15];
    const float* vf_b1  = (const float*)d_in[16];
    const float* vf_lng = (const float*)d_in[17];
    const float* vf_lnb = (const float*)d_in[18];
    const float* vf_w2  = (const float*)d_in[19];
    const float* vf_b2  = (const float*)d_in[20];
    const float* pred_w = (const float*)d_in[21];
    const float* pred_b = (const float*)d_in[22];
    const float* bn_g   = (const float*)d_in[23];
    const float* bn_b   = (const float*)d_in[24];
    const float* dec_w  = (const float*)d_in[25];
    const float* dec_b  = (const float*)d_in[26];
    const float* hx     = (const float*)d_in[27];
    const float* cx     = (const float*)d_in[28];
    const int*   tokens = (const int*)d_in[29];

    float* out = (float*)d_out;
    const size_t OFS_DEC = 0;
    const size_t OFS_H   = (size_t)T * B * NT;                 // 131,072,000
    const size_t OFS_C   = OFS_H + (size_t)2 * B * H;
    const size_t OFS_V   = OFS_C + (size_t)2 * B * H;

    // ---- workspace bump allocator (256B aligned)
    char* base = (char*)d_ws; size_t off = 0;
    auto alloc = [&](size_t bytes) -> void* {
        void* p = base + off; off += (bytes + 255) & ~(size_t)255; return p;
    };
    typedef unsigned short u16;
    u16* wb_l0_wih = (u16*)alloc((size_t)4 * H * NI * 2);        // [4096 x 512]
    u16* wb_l0_whh = (u16*)alloc((size_t)4 * H * H * 2);         // [4096 x 1024]
    u16* wb_rc_wih = (u16*)alloc((size_t)4 * H * H * 2);
    u16* wb_rc_whh = (u16*)alloc((size_t)4 * H * H * 2);
    u16* wb_head   = (u16*)alloc((size_t)2 * NI * 3 * H * 2);    // [1024 x 3072] pi;vf
    u16* wb_pred   = (u16*)alloc((size_t)NI * H * 2);            // [512 x 1024]
    u16* wb_dec    = (u16*)alloc((size_t)NT * NI * 2);           // [32000 x 512]
    u16* emb_bf    = (u16*)alloc((size_t)T * B * NI * 2);        // [4096 x 512]
    u16* state_bf  = (u16*)alloc((size_t)B * 3 * H * 2);
    u16* h0_bf     = (u16*)alloc((size_t)B * H * 2);
    u16* h1_bf0    = (u16*)alloc((size_t)B * H * 2);
    u16* h1_bf1    = (u16*)alloc((size_t)B * H * 2);
    u16* seq_bf    = (u16*)alloc((size_t)T * B * H * 2);         // [4096 x 1024]
    u16* ybn_bf    = (u16*)alloc((size_t)T * B * NI * 2);        // [4096 x 512]
    float* bias_l0   = (float*)alloc((size_t)4 * H * 4);
    float* bias_rc   = (float*)alloc((size_t)4 * H * 4);
    float* bias_head = (float*)alloc((size_t)2 * NI * 4);
    float* g_buf  = (float*)alloc((size_t)B * 4 * H * 4);
    float* y_buf  = (float*)alloc((size_t)B * 2 * NI * 4);
    float* h0_f   = (float*)alloc((size_t)B * H * 4);
    float* c0_f   = (float*)alloc((size_t)B * H * 4);
    float* h1_f0  = (float*)alloc((size_t)B * H * 4);
    float* h1_f1  = (float*)alloc((size_t)B * H * 4);
    float* c1_f0  = (float*)alloc((size_t)B * H * 4);
    float* c1_f1  = (float*)alloc((size_t)B * H * 4);
    float* hr_f   = (float*)alloc((size_t)B * H * 4);
    float* cr_f   = (float*)alloc((size_t)B * H * 4);
    float* si_f   = (float*)alloc((size_t)B * H * 4);
    float* stk_f  = (float*)alloc((size_t)B * 2 * H * 4);
    int*   slen   = (int*)alloc((size_t)B * 4);

    const int TPB = 256;
    auto cvt = [&](const float* src, u16* dst, int n) {
        cvt_bf16_kernel<<<cdiv(n, TPB), TPB, 0, stream>>>(src, dst, n);
    };

    // ---- setup: weights -> bf16 (fits in L2; halves per-step weight traffic)
    cvt(l0_wih, wb_l0_wih, 4 * H * NI);
    cvt(l0_whh, wb_l0_whh, 4 * H * H);
    cvt(rc_wih, wb_rc_wih, 4 * H * H);
    cvt(rc_whh, wb_rc_whh, 4 * H * H);
    cvt(pi_w1, wb_head, NI * 3 * H);
    cvt(vf_w1, wb_head + (size_t)NI * 3 * H, NI * 3 * H);
    cvt(pred_w, wb_pred, NI * H);
    cvt(dec_w, wb_dec, NT * NI);
    add2_kernel<<<cdiv(4 * H, TPB), TPB, 0, stream>>>(l0_bih, l0_bhh, bias_l0, 4 * H);
    add2_kernel<<<cdiv(4 * H, TPB), TPB, 0, stream>>>(rc_bih, rc_bhh, bias_rc, 4 * H);
    copyf_kernel<<<cdiv(NI, TPB), TPB, 0, stream>>>(pi_b1, bias_head, NI);
    copyf_kernel<<<cdiv(NI, TPB), TPB, 0, stream>>>(vf_b1, bias_head + NI, NI);
    embed_kernel<<<cdiv(T * B * NI, TPB), TPB, 0, stream>>>(emb_W, tokens, emb_bf, T * B, NI);
    init_state_kernel<<<cdiv(B * H, TPB), TPB, 0, stream>>>(hx, cx, h0_f, c0_f, h1_f0, c1_f0,
                                                            h0_bf, h1_bf0, stk_f, slen);

    float* h1f[2] = { h1_f0, h1_f1 };
    float* c1f[2] = { c1_f0, c1_f1 };
    u16*   h1b[2] = { h1_bf0, h1_bf1 };

    // ---- sequential scan (L2-bandwidth bound on bf16 weights; NTW=1 for max spread)
    for (int t = 0; t < T; ++t) {
        const int cur = t & 1, nxt = cur ^ 1;
        // layer-0 LSTM: g = emb_t*Wih^T + h0*Whh^T + b
        gemm_bf16_wmma<0, 1><<<dim3(4 * H / 64, B / 16), 128, 0, stream>>>(
            emb_bf + (size_t)t * B * NI, wb_l0_wih, NI,
            h0_bf, wb_l0_whh, H, bias_l0, g_buf, nullptr, 4 * H, nullptr, nullptr, 0.f);
        lstm_cell_kernel<<<cdiv(B * H, TPB), TPB, 0, stream>>>(g_buf, c0_f, h0_f, c0_f, h0_bf);
        // recurrent LSTM: gr = hi*rcWih^T + htm1*rcWhh^T + b -> hr, cr
        gemm_bf16_wmma<0, 1><<<dim3(4 * H / 64, B / 16), 128, 0, stream>>>(
            h0_bf, wb_rc_wih, H, h1b[cur], wb_rc_whh, H,
            bias_rc, g_buf, nullptr, 4 * H, nullptr, nullptr, 0.f);
        lstm_cell_kernel<<<cdiv(B * H, TPB), TPB, 0, stream>>>(g_buf, c1f[cur], hr_f, cr_f, nullptr);
        // state = [htm1 | si | hi]
        build_state_kernel<<<cdiv(B * 3 * H, TPB), TPB, 0, stream>>>(h1f[cur], h0_f, stk_f, slen,
                                                                     state_bf, si_f);
        // both MLP heads in one GEMM: [32 x 3072] x [3072 x 1024]
        gemm_bf16_wmma<0, 1><<<dim3(2 * NI / 64, B / 16), 128, 0, stream>>>(
            state_bf, wb_head, 3 * H, nullptr, nullptr, 0,
            bias_head, y_buf, nullptr, 2 * NI, nullptr, nullptr, 0.f);
        // LN/tanh heads, masked argmax, merge/push, stack + state update
        finalize_kernel<<<B, 256, 0, stream>>>(
            y_buf, hr_f, cr_f, si_f,
            pi_lng, pi_lnb, pi_w2, pi_b2, vf_lng, vf_lnb, vf_w2, vf_b2,
            stk_f, slen, h1f[nxt], c1f[nxt], h1b[nxt], seq_bf, out + OFS_V, t);
    }
    const int fin = T & 1;   // index holding final h1/c1 after last swap

    // ---- tail: predictor (fused BN+tanh epilogue) then decoder GEMM into d_out
    // NTW=4: each wave computes 16x64, A fragment reused 4x (compute-heavy GEMMs)
    const float inv_std = 1.0f / sqrtf(1.0f + 1e-5f);
    gemm_bf16_wmma<1, 4><<<dim3(NI / 256, T * B / 16), 128, 0, stream>>>(
        seq_bf, wb_pred, H, nullptr, nullptr, 0,
        pred_b, nullptr, ybn_bf, NI, bn_g, bn_b, inv_std);
    gemm_bf16_wmma<0, 4><<<dim3(NT / 256, T * B / 16), 128, 0, stream>>>(
        ybn_bf, wb_dec, NI, nullptr, nullptr, 0,
        dec_b, out + OFS_DEC, nullptr, NT, nullptr, nullptr, 0.f);
    copy_hc_kernel<<<cdiv(B * H, TPB), TPB, 0, stream>>>(h0_f, h1f[fin], c0_f, c1f[fin],
                                                         out + OFS_H, out + OFS_C);
}
